// RGCNModel3_4080218931586
// MI455X (gfx1250) — matrix operand order repaired and checked by reading, not compiled, not run on hardware
//
#include <hip/hip_runtime.h>
#include <hip/hip_bf16.h>
#include <cmath>

typedef __attribute__((ext_vector_type(16))) _Float16 v16h;
typedef __attribute__((ext_vector_type(8)))  _Float16 v8h;
typedef __attribute__((ext_vector_type(8)))  float    v8f;

// explicit global-address-space pointer types (force global_* VMEM, not flat_*)
typedef const __attribute__((address_space(1))) _Float16* gch_p;
typedef const __attribute__((address_space(1))) v8h*      gcv8h_p;
typedef const __attribute__((address_space(1))) float*    gcf_p;
typedef __attribute__((address_space(1))) float*          gf_p;
typedef __attribute__((address_space(1))) _Float16*       gh_p;

union Frag16 { v16h v; v8h h[2]; };

// ---------------------------------------------------------------- utilities
__global__ void zero_f32_k(float* __restrict__ p, long n) {
  long i = (long)blockIdx.x * blockDim.x + threadIdx.x;
  long st = (long)gridDim.x * blockDim.x;
  for (; i < n; i += st) p[i] = 0.f;
}

__global__ void f32_to_f16_k(const float* __restrict__ in, _Float16* __restrict__ out, long n) {
  long i = (long)blockIdx.x * blockDim.x + threadIdx.x;
  long st = (long)gridDim.x * blockDim.x;
  for (; i < n; i += st) out[i] = (_Float16)in[i];
}

// W[r] = sum_b comb[r,b] * basis[b]  -> f16 [in*out]
__global__ void build_relW_k(const float* __restrict__ comb, const float* __restrict__ basis,
                             int r, int nb, long dd, _Float16* __restrict__ out) {
  long i = (long)blockIdx.x * blockDim.x + threadIdx.x;
  long st = (long)gridDim.x * blockDim.x;
  for (; i < dd; i += st) {
    float acc = 0.f;
    for (int b = 0; b < nb; ++b) acc += comb[r * nb + b] * basis[(long)b * dd + i];
    out[i] = (_Float16)acc;
  }
}

// S[dst] += x[src] for edges of relation `rel`
__global__ void scatter_rel_k(const float* __restrict__ x, const int* __restrict__ src,
                              const int* __restrict__ dst, const int* __restrict__ et,
                              int rel, int E, int dim, float* __restrict__ S) {
  int e = blockIdx.x;
  if (e >= E || et[e] != rel) return;
  const float* xp = x + (size_t)src[e] * dim;
  float* sp = S + (size_t)dst[e] * dim;
  for (int c = threadIdx.x; c < dim; c += blockDim.x) atomicAdd(sp + c, xp[c]);
}

// one wave per (edge, head): score + weighted-V / z scatter
__global__ void attn_edges_k(const float* __restrict__ Q, const float* __restrict__ K,
                             const float* __restrict__ V, const int* __restrict__ src,
                             const int* __restrict__ dst, int E, int dim, int H,
                             float inv_scale, float* __restrict__ attnb, float* __restrict__ z) {
  int wave = blockIdx.x * (blockDim.x >> 5) + (threadIdx.x >> 5);
  int lane = threadIdx.x & 31;
  long total = (long)E * H;
  if (wave >= total) return;
  int e = wave / H, h = wave % H;
  int dh = dim / H;
  int s = src[e], d = dst[e];
  const float* kp = K + (size_t)s * dim + h * dh;
  const float* qp = Q + (size_t)d * dim + h * dh;
  const float* vp = V + (size_t)s * dim + h * dh;
  float acc = 0.f;
  for (int c = lane; c < dh; c += 32) acc += kp[c] * qp[c];
  for (int off = 16; off > 0; off >>= 1) acc += __shfl_xor(acc, off, 32);
  float sc = __expf(fminf(fmaxf(acc * inv_scale, -10.f), 10.f));
  if (lane == 0) atomicAdd(&z[(size_t)d * H + h], sc);
  float* ap = attnb + (size_t)d * dim + h * dh;
  for (int c = lane; c < dh; c += 32) atomicAdd(ap + c, vp[c] * sc);
}

// out16[n, col_off + c] = attn[n,c] / (z[n,h] + 1e-6)
__global__ void attn_finalize_k(const float* __restrict__ attnb, const float* __restrict__ z,
                                int N, int dim, int H, _Float16* __restrict__ out16,
                                int ld16, int col_off) {
  long idx = (long)blockIdx.x * blockDim.x + threadIdx.x;
  long total = (long)N * dim;
  if (idx >= total) return;
  int n = (int)(idx / dim);
  int c = (int)(idx % dim);
  int h = c / (dim / H);
  float v = attnb[idx] / (z[(size_t)n * H + h] + 1e-6f);
  out16[(size_t)n * ld16 + col_off + c] = (_Float16)v;
}

// ------------------------------------------------------------- WMMA GEMM
// C[M,N] (+)= A[M,K](f16) @ B[K,N](f16); optional bias+relu fuse; optional f16 copy.
// Block: 128 threads = 4 waves; block tile 64x128; wave tile 32x64 (2x4 WMMA tiles).
// OOB A rows are clamped to M-1 (their outputs are never stored), keeping the
// K-loop branch-free. All memory traffic uses explicit AS(1) pointers so it
// lowers to global_load_b128/global_store (LOADcnt/STOREcnt only, no flat path).
__global__ __launch_bounds__(128)
void wmma_gemm_k(const _Float16* __restrict__ A, int lda,
                 const _Float16* __restrict__ B, int ldb,
                 float* __restrict__ C, int ldc,
                 _Float16* __restrict__ C16, int ldc16,
                 const float* __restrict__ bias,
                 int M, int N, int K, int accum, int fuse) {
  const int lane = threadIdx.x & 31;
  const int wave = threadIdx.x >> 5;
  const int bm = blockIdx.x * 64 + (wave >> 1) * 32;
  const int bn = blockIdx.y * 128 + (wave & 1) * 64;
  const int half = lane >> 4;
  const int l16 = lane & 15;

  gch_p Ag = (gch_p)A;
  gch_p Bg = (gch_p)B;
  gf_p  Cg = (gf_p)C;
  gh_p  C16g = (gh_p)C16;
  gcf_p biasg = (gcf_p)bias;

  v8f c[2][4];
#pragma unroll
  for (int i = 0; i < 2; ++i)
#pragma unroll
    for (int j = 0; j < 4; ++j) {
      v8f t = {};
      if (accum) {
#pragma unroll
        for (int e = 0; e < 8; ++e) {
          int m = bm + i * 16 + half * 8 + e;
          int n = bn + j * 16 + l16;
          if (m < M) t[e] = Cg[(size_t)m * ldc + n];
        }
      }
      c[i][j] = t;
    }

  // A frag: lane l16 = row; half selects K sub-chunks {0..7,16..23} vs {8..15,24..31}
  gch_p aptr[2];
#pragma unroll
  for (int i = 0; i < 2; ++i) {
    int m = bm + i * 16 + l16;
    if (m >= M) m = M - 1;  // clamp: result rows >= M are never stored
    aptr[i] = Ag + (size_t)m * lda + half * 8;
  }
  // B frag: lane l16 = N col; half selects K sub-chunks {0..7,16..23} vs {8..15,24..31}
  gch_p bptr[4];
#pragma unroll
  for (int j = 0; j < 4; ++j) bptr[j] = Bg + (size_t)(half * 8) * ldb + bn + j * 16 + l16;

  const size_t bstep = (size_t)32 * ldb;
  for (int kk = 0; kk < K; kk += 32) {
    Frag16 a[2], b[4];
#pragma unroll
    for (int i = 0; i < 2; ++i) {
      a[i].h[0] = *(gcv8h_p)(aptr[i]);
      a[i].h[1] = *(gcv8h_p)(aptr[i] + 16);
      aptr[i] += 32;
    }
#pragma unroll
    for (int j = 0; j < 4; ++j) {
#pragma unroll
      for (int s = 0; s < 8; ++s) {
        b[j].v[s] = bptr[j][(size_t)s * ldb];
        b[j].v[8 + s] = bptr[j][(size_t)(16 + s) * ldb];
      }
      bptr[j] += bstep;
    }
#pragma unroll
    for (int i = 0; i < 2; ++i)
#pragma unroll
      for (int j = 0; j < 4; ++j)
        c[i][j] = __builtin_amdgcn_wmma_f32_16x16x32_f16(
            false, a[i].v, false, b[j].v, (short)0, c[i][j], false, false);
  }

#pragma unroll
  for (int i = 0; i < 2; ++i)
#pragma unroll
    for (int j = 0; j < 4; ++j) {
      int n = bn + j * 16 + l16;
      float bv = fuse ? biasg[n] : 0.f;
#pragma unroll
      for (int e = 0; e < 8; ++e) {
        int m = bm + i * 16 + half * 8 + e;
        if (m < M) {
          float v = c[i][j][e];
          if (fuse) v = fmaxf(v + bv, 0.f);
          Cg[(size_t)m * ldc + n] = v;
          if (C16) C16g[(size_t)m * ldc16 + n] = (_Float16)v;
        }
      }
    }
}

__global__ void mlp3_sigmoid_k(const float* __restrict__ m2, const float* __restrict__ w,
                               const float* __restrict__ b, float* __restrict__ out,
                               int N, int K) {
  int n = blockIdx.x * blockDim.x + threadIdx.x;
  if (n >= N) return;
  float acc = b[0];
  const float* row = m2 + (size_t)n * K;
  for (int j = 0; j < K; ++j) acc += row[j] * w[j];
  out[n] = 1.f / (1.f + __expf(-acc));
}

// ------------------------------------------------------------- host side
namespace {
struct QKVSpec { const float* comb; const float* basis; const float* bias; float* out; };

void launch_attention(const float* x, int dim, const QKVSpec* parts,
                      const int* src, const int* dst, const int* et, int N, int E,
                      float* S, _Float16* Sh, _Float16* W16,
                      float* attnb, float* z,
                      _Float16* out16, int ld16, int col_off, hipStream_t s) {
  const int H = 4, R = 6, NB = 6;
  long nd = (long)N * dim;
  long dd = (long)dim * dim;
  dim3 gg((N + 63) / 64, dim / 128);
  for (int r = 0; r < R; ++r) {
    zero_f32_k<<<2048, 256, 0, s>>>(S, nd);
    scatter_rel_k<<<E, 256, 0, s>>>(x, src, dst, et, r, E, dim, S);
    f32_to_f16_k<<<2048, 256, 0, s>>>(S, Sh, nd);
    for (int p = 0; p < 3; ++p) {
      build_relW_k<<<2048, 256, 0, s>>>(parts[p].comb, parts[p].basis, r, NB, dd, W16);
      wmma_gemm_k<<<gg, 128, 0, s>>>(Sh, dim, W16, dim, parts[p].out, dim, nullptr, 0,
                                     parts[p].bias, N, dim, dim, r > 0 ? 1 : 0,
                                     (r == R - 1) ? 1 : 0);
    }
  }
  zero_f32_k<<<2048, 256, 0, s>>>(attnb, nd);
  zero_f32_k<<<64, 256, 0, s>>>(z, (long)N * H);
  long tasks = (long)E * H;
  float inv_scale = 1.f / sqrtf((float)(dim / H));
  attn_edges_k<<<(int)((tasks + 7) / 8), 256, 0, s>>>(parts[0].out, parts[1].out, parts[2].out,
                                                      src, dst, E, dim, H, inv_scale, attnb, z);
  attn_finalize_k<<<(int)((nd + 255) / 256), 256, 0, s>>>(attnb, z, N, dim, H, out16, ld16, col_off);
}
}  // namespace

extern "C" void kernel_launch(void* const* d_in, const int* in_sizes, int n_in,
                              void* d_out, int out_size, void* d_ws, size_t ws_size,
                              hipStream_t stream) {
  const int IN_FEATS = 768, GS = 512, H = 4;
  const int N = in_sizes[0] / IN_FEATS;
  const int E = in_sizes[1];

  const float* features = (const float*)d_in[0];
  const int* src = (const int*)d_in[1];
  const int* dst = (const int*)d_in[2];
  const int* et  = (const int*)d_in[3];
  const float* merge1_w = (const float*)d_in[4];
  const float* merge1_b = (const float*)d_in[5];
  // att1 q/k/v: (comb, basis, bias) triplets at 6..14
  const float* merge2_w = (const float*)d_in[15];
  const float* merge2_b = (const float*)d_in[16];
  // att2 q/k/v at 17..25
  const float* mlp1_w = (const float*)d_in[26];
  const float* mlp1_b = (const float*)d_in[27];
  const float* mlp2_w = (const float*)d_in[28];
  const float* mlp2_b = (const float*)d_in[29];
  const float* mlp3_w = (const float*)d_in[30];
  const float* mlp3_b = (const float*)d_in[31];

  // ---- carve workspace
  size_t off = 0;
  auto carve = [&](size_t bytes) -> void* {
    void* p = (char*)d_ws + off;
    off += (bytes + 255) & ~(size_t)255;
    return p;
  };
  _Float16* feat16   = (_Float16*)carve((size_t)N * IN_FEATS * 2);
  _Float16* w_m1_16  = (_Float16*)carve((size_t)IN_FEATS * 1024 * 2);
  _Float16* w_m2_16  = (_Float16*)carve((size_t)1024 * 512 * 2);
  _Float16* w_p1_16  = (_Float16*)carve((size_t)1536 * 512 * 2);
  _Float16* w_p2_16  = (_Float16*)carve((size_t)512 * 128 * 2);
  _Float16* W16      = (_Float16*)carve((size_t)1024 * 1024 * 2);
  float*    x1       = (float*)carve((size_t)N * 1024 * 4);
  float*    x2       = (float*)carve((size_t)N * 512 * 4);
  float*    S        = (float*)carve((size_t)N * 1024 * 4);
  _Float16* Sh       = (_Float16*)carve((size_t)N * 1024 * 2);
  float*    Qb       = (float*)carve((size_t)N * 1024 * 4);
  float*    Kb       = (float*)carve((size_t)N * 1024 * 4);
  float*    Vb       = (float*)carve((size_t)N * 1024 * 4);
  float*    attnb    = (float*)carve((size_t)N * 1024 * 4);
  float*    z        = (float*)carve((size_t)N * H * 4);
  _Float16* cat16    = (_Float16*)carve((size_t)N * 1536 * 2);
  float*    m1       = (float*)carve((size_t)N * 512 * 4);
  _Float16* m1h      = (_Float16*)carve((size_t)N * 512 * 2);
  float*    m2       = (float*)carve((size_t)N * 128 * 4);
  (void)ws_size; (void)n_in; (void)out_size;

  // ---- static f16 conversions
  f32_to_f16_k<<<2048, 256, 0, stream>>>(features, feat16, (long)N * IN_FEATS);
  f32_to_f16_k<<<2048, 256, 0, stream>>>(merge1_w, w_m1_16, (long)IN_FEATS * 1024);
  f32_to_f16_k<<<2048, 256, 0, stream>>>(merge2_w, w_m2_16, (long)1024 * 512);
  f32_to_f16_k<<<2048, 256, 0, stream>>>(mlp1_w, w_p1_16, (long)1536 * 512);
  f32_to_f16_k<<<512, 256, 0, stream>>>(mlp2_w, w_p2_16, (long)512 * 128);

  // ---- merge1: x1 = relu(features @ W + b)  [N,1024]
  {
    dim3 g((N + 63) / 64, 1024 / 128);
    wmma_gemm_k<<<g, 128, 0, stream>>>(feat16, IN_FEATS, w_m1_16, 1024, x1, 1024,
                                       nullptr, 0, merge1_b, N, 1024, IN_FEATS, 0, 1);
  }

  // ---- att1 (dim 1024) -> cat16[:, 0:1024]
  {
    QKVSpec parts[3] = {
      {(const float*)d_in[6],  (const float*)d_in[7],  (const float*)d_in[8],  Qb},
      {(const float*)d_in[9],  (const float*)d_in[10], (const float*)d_in[11], Kb},
      {(const float*)d_in[12], (const float*)d_in[13], (const float*)d_in[14], Vb}};
    launch_attention(x1, 2 * GS, parts, src, dst, et, N, E, S, Sh, W16, attnb, z,
                     cat16, 1536, 0, stream);
  }

  // ---- merge2: x2 = relu(att1_out @ W + b)  [N,512], A = cat16[:, :1024], lda=1536
  {
    dim3 g((N + 63) / 64, 512 / 128);
    wmma_gemm_k<<<g, 128, 0, stream>>>(cat16, 1536, w_m2_16, 512, x2, 512,
                                       nullptr, 0, merge2_b, N, 512, 1024, 0, 1);
  }

  // ---- att2 (dim 512) -> cat16[:, 1024:1536]
  {
    QKVSpec parts[3] = {
      {(const float*)d_in[17], (const float*)d_in[18], (const float*)d_in[19], Qb},
      {(const float*)d_in[20], (const float*)d_in[21], (const float*)d_in[22], Kb},
      {(const float*)d_in[23], (const float*)d_in[24], (const float*)d_in[25], Vb}};
    launch_attention(x2, GS, parts, src, dst, et, N, E, S, Sh, W16, attnb, z,
                     cat16, 1536, 1024, stream);
  }

  // ---- mlp1: m1 = relu(cat @ W + b) [N,512] (+f16 copy)
  {
    dim3 g((N + 63) / 64, 512 / 128);
    wmma_gemm_k<<<g, 128, 0, stream>>>(cat16, 1536, w_p1_16, 512, m1, 512,
                                       m1h, 512, mlp1_b, N, 512, 1536, 0, 1);
  }
  // ---- mlp2: m2 = relu(m1 @ W + b) [N,128]
  {
    dim3 g((N + 63) / 64, 128 / 128);
    wmma_gemm_k<<<g, 128, 0, stream>>>(m1h, 512, w_p2_16, 128, m2, 128,
                                       nullptr, 0, mlp2_b, N, 128, 512, 0, 1);
  }
  // ---- mlp3 + sigmoid
  mlp3_sigmoid_k<<<(N + 255) / 256, 256, 0, stream>>>(m2, mlp3_w, mlp3_b, (float*)d_out, N, 128);
}
